// LSTM_80616536146186
// MI455X (gfx1250) — matrix operand order repaired and checked by reading, not compiled, not run on hardware
//
#include <hip/hip_runtime.h>
#include <hip/hip_bf16.h>

#define BSZ  256
#define TLEN 2048
#define HDIM 128

typedef __attribute__((ext_vector_type(16))) _Float16 v16h;
typedef __attribute__((ext_vector_type(8)))  _Float16 v8h;
typedef __attribute__((ext_vector_type(8)))  float    v8f;

#if defined(__has_builtin) && __has_builtin(__builtin_amdgcn_tanhf)
__device__ __forceinline__ float ftanh(float v) {
    return __builtin_amdgcn_tanhf(v);        // v_tanh_f32 (gfx1250 TRANS op)
}
#else
#define LOG2E 1.44269504088896341f
__device__ __forceinline__ float ftanh(float v) {
    const float e = __builtin_amdgcn_exp2f(v * (2.0f * LOG2E));
    return 1.0f - 2.0f * __builtin_amdgcn_rcpf(e + 1.0f);
}
#endif

// Sigmoid-gate weights are pre-scaled by 0.5, so exactly:
// sigmoid(x) = 0.5*tanh(x/2) + 0.5  ->  1 TRANS + 1 FMA.
__device__ __forceinline__ float fsig_pre(float v) {
    return 0.5f * ftanh(v) + 0.5f;
}

// One workgroup = 16 batch rows, 8 waves (wave32). Wave w owns H-cols
// [16w, 16w+16) and gate cols 16w + {0,128,256,384}. W_hh lives in VGPRs
// as f16 WMMA B-fragments; h is double-buffered in LDS as f16. The FC
// head is computed from the published h rows, spread across all waves.
__global__ __launch_bounds__(256) void lstm_wmma_kernel(
    const float* __restrict__ x,      // [B, T]
    const float* __restrict__ W_ih,   // [4H, 1]
    const float* __restrict__ W_hh,   // [4H, H]
    const float* __restrict__ b_ih,   // [4H]
    const float* __restrict__ b_hh,   // [4H]
    const float* __restrict__ fc_w,   // [1, H]
    const float* __restrict__ fc_b,   // [1]
    float* __restrict__ out)          // [B*T] ++ [B*H] ++ [B*H]
{
    __shared__ _Float16 hbuf[2][16][HDIM];   // 8 KB, double-buffered h state

    const int tid  = threadIdx.x;
    const int wid  = tid >> 5;
    const int lane = tid & 31;
    const int nh   = lane & 15;   // N (column) within 16x16 tile
    const int hi   = lane >> 4;   // lane half (selects K group / M+8)
    const int m0   = blockIdx.x * 16;

    // zero hbuf[0] (initial h state): 4096 B, 16 B per thread
    ((uint4*)&hbuf[0][0][0])[tid] = uint4{0u, 0u, 0u, 0u};

    // gate scale: i,f,o pre-scaled by 0.5 for tanh-based sigmoid; g raw.
    const float gscl[4] = { 0.5f, 0.5f, 1.0f, 0.5f };

    // ---- per-lane constants -------------------------------------------
    float wih[4], bsum[4];
#pragma unroll
    for (int g = 0; g < 4; ++g) {
        const int col = g * HDIM + wid * 16 + nh;
        wih[g]  = W_ih[col] * gscl[g];
        bsum[g] = (b_ih[col] + b_hh[col]) * gscl[g];
    }
    const float fcb = fc_b[0];

    // FC weights for this lane's 8-column chunk (f16, 4 VGPRs)
    v8h fcw8;
#pragma unroll
    for (int j = 0; j < 8; ++j)
        fcw8[j] = (_Float16)fc_w[nh * 8 + j];

    // ---- W_hh -> f16 B fragments (pre-scaled), resident in VGPRs ------
    // B (32x16, K x N) layout: lane half 0 holds K=0..7,16..23,
    // lanes 16-31 hold K=8..15,24..31; N = lane&15.
    v16h Bf[4][4];
#pragma unroll
    for (int g = 0; g < 4; ++g) {
        const int row = g * HDIM + wid * 16 + nh;   // W_hh row = gate col
        const float* wr = W_hh + row * HDIM;
#pragma unroll
        for (int k = 0; k < 4; ++k) {
            const int kb = k * 32 + hi * 8;
#pragma unroll
            for (int j = 0; j < 16; ++j)
                Bf[g][k][j] = (_Float16)(wr[kb + j + (j >= 8 ? 8 : 0)] * gscl[g]);
        }
    }

    // ---- state registers ----------------------------------------------
    float c[8], hreg[8];
#pragma unroll
    for (int r = 0; r < 8; ++r) { c[r] = 0.0f; hreg[r] = 0.0f; }

    float xcur[8];
#pragma unroll
    for (int r = 0; r < 8; ++r)
        xcur[r] = x[(m0 + r + 8 * hi) * TLEN + 0];

    __syncthreads();

    // ---- one LSTM step; p is a compile-time buffer parity --------------
    auto step = [&](int t, int p) __attribute__((always_inline)) {
        // prefetch next x (hidden behind WMMAs)
        float xnext[8];
        if (t + 1 < TLEN) {
#pragma unroll
            for (int r = 0; r < 8; ++r)
                xnext[r] = x[(m0 + r + 8 * hi) * TLEN + (t + 1)];
        } else {
#pragma unroll
            for (int r = 0; r < 8; ++r) xnext[r] = 0.0f;
        }

        // FC head for step t-1 from the published h rows in hbuf[p].
        // Wave w handles rows {2w, 2w+1}: hi selects row, nh the 8-col chunk.
        if (t > 0) {
            const int row = 2 * wid + hi;
            const v8h hv = *(const v8h*)&hbuf[p][row][nh * 8];
            float s = 0.0f;
#pragma unroll
            for (int j = 0; j < 8; ++j)
                s = fmaf((float)hv[j], (float)fcw8[j], s);
#pragma unroll
            for (int off = 1; off < 16; off <<= 1)
                s += __shfl_xor(s, off, 32);
            if (nh == 0)
                out[(m0 + row) * TLEN + (t - 1)] = ftanh(s + fcb);
        }

        // A fragments from LDS (16-bit A 16x32 layout):
        // lane half 0: K = kb+0..7 and kb+16..23 ; half 1: +8
        v16h A[4];
#pragma unroll
        for (int k = 0; k < 4; ++k) {
            const v8h lo = *(const v8h*)&hbuf[p][nh][k * 32 + hi * 8];
            const v8h hc = *(const v8h*)&hbuf[p][nh][k * 32 + 16 + hi * 8];
#pragma unroll
            for (int j = 0; j < 8; ++j) { A[k][j] = lo[j]; A[k][j + 8] = hc[j]; }
        }

        // accumulators init = x_t * W_ih + (b_ih + b_hh)   (pre-scaled)
        v8f acc[4];
#pragma unroll
        for (int g = 0; g < 4; ++g) {
#pragma unroll
            for (int r = 0; r < 8; ++r)
                acc[g][r] = xcur[r] * wih[g] + bsum[g];
        }

        // 16 WMMAs: 4 K-chunks x 4 gates, D = A*B + C (f32 acc)
#pragma unroll
        for (int k = 0; k < 4; ++k) {
#pragma unroll
            for (int g = 0; g < 4; ++g) {
                acc[g] = __builtin_amdgcn_wmma_f32_16x16x32_f16(
                    false, A[k], false, Bf[g][k], (short)0, acc[g],
                    false, false);
            }
        }

        // gates + state update; publish h (f16) for next step
#pragma unroll
        for (int r = 0; r < 8; ++r) {
            const float ig = fsig_pre(acc[0][r]);
            const float fg = fsig_pre(acc[1][r]);
            const float gg = ftanh(acc[2][r]);
            const float og = fsig_pre(acc[3][r]);
            c[r] = fg * c[r] + ig * gg;
            const float hn = og * ftanh(c[r]);
            hreg[r] = hn;
            hbuf[1 - p][r + 8 * hi][wid * 16 + nh] = (_Float16)hn;
        }

#pragma unroll
        for (int r = 0; r < 8; ++r) xcur[r] = xnext[r];

        __syncthreads();
    };

    // ---- time loop, unrolled x2 so buffer parity is compile-time -------
    for (int t = 0; t < TLEN; t += 2) {
        step(t, 0);
        step(t + 1, 1);
    }

    // final FC output for t = TLEN-1 (h lives in hbuf[TLEN & 1] = hbuf[0])
    {
        const int row = 2 * wid + hi;
        const v8h hv = *(const v8h*)&hbuf[0][row][nh * 8];
        float s = 0.0f;
#pragma unroll
        for (int j = 0; j < 8; ++j)
            s = fmaf((float)hv[j], (float)fcw8[j], s);
#pragma unroll
        for (int off = 1; off < 16; off <<= 1)
            s += __shfl_xor(s, off, 32);
        if (nh == 0)
            out[(m0 + row) * TLEN + (TLEN - 1)] = ftanh(s + fcb);
    }

    // hT, cT ([1,B,H] each), appended after out
    float* hT = out + (size_t)BSZ * TLEN;
    float* cT = hT + (size_t)BSZ * HDIM;
#pragma unroll
    for (int r = 0; r < 8; ++r) {
        const int b   = m0 + r + 8 * hi;
        const int col = wid * 16 + nh;
        hT[b * HDIM + col] = hreg[r];
        cT[b * HDIM + col] = c[r];
    }
}

extern "C" void kernel_launch(void* const* d_in, const int* in_sizes, int n_in,
                              void* d_out, int out_size, void* d_ws, size_t ws_size,
                              hipStream_t stream) {
    (void)in_sizes; (void)n_in; (void)d_ws; (void)ws_size; (void)out_size;
    const float* x    = (const float*)d_in[0];
    const float* W_ih = (const float*)d_in[1];
    const float* W_hh = (const float*)d_in[2];
    const float* b_ih = (const float*)d_in[3];
    const float* b_hh = (const float*)d_in[4];
    const float* fc_w = (const float*)d_in[5];
    const float* fc_b = (const float*)d_in[6];
    float* out = (float*)d_out;

    dim3 grid(BSZ / 16);   // 16 workgroups, one per 16 batch rows
    dim3 block(256);       // 8 wave32 waves
    lstm_wmma_kernel<<<grid, block, 0, stream>>>(x, W_ih, W_hh, b_ih, b_hh,
                                                 fc_w, fc_b, out);
}
